// TransformerDecoderLayer_5463198401294
// MI455X (gfx1250) — compile-verified, matrix operand order checked by reading, not run on hardware
//
#include <hip/hip_runtime.h>

// Problem constants (match reference)
constexpr int cN = 4, cS = 1024, cD = 1024, cH = 16, cHD = 64, cHID = 4096;

typedef __attribute__((ext_vector_type(16))) __bf16 v16bf;
typedef __attribute__((ext_vector_type(8)))  __bf16 v8bf;
typedef __attribute__((ext_vector_type(8)))  float  v8f;

__device__ __forceinline__ __bf16 f2bf(float f) { return (__bf16)f; }

__device__ __forceinline__ v8f wmma_bf16(v16bf a, v16bf b, v8f c) {
  return __builtin_amdgcn_wmma_f32_16x16x32_bf16(false, a, false, b, (short)0, c, false, false);
}

__device__ __forceinline__ v16bf cat8(v8bf lo, v8bf hi) {
  return __builtin_shufflevector(lo, hi, 0, 1, 2, 3, 4, 5, 6, 7, 8, 9, 10, 11, 12, 13, 14, 15);
}

// A-fragment (16x32 bf16) from a row-major bf16 matrix: lane l -> row l&15, K base (l>>4)*8.
__device__ __forceinline__ v16bf load_afrag_bf16(const __bf16* __restrict__ base, int ld, int lane) {
  const __bf16* p = base + (size_t)(lane & 15) * ld + ((lane >> 4) * 8);
  v8bf lo = *(const v8bf*)p;
  v8bf hi = *(const v8bf*)(p + 16);
  return cat8(lo, hi);
}

// ---------------- fp32 -> bf16 bulk convert (8 elems/thread, coalesced) ----------------
__global__ __launch_bounds__(256) void cvt_bf16_kernel(const float* __restrict__ in,
                                                       __bf16* __restrict__ out) {
  const size_t i = ((size_t)blockIdx.x * 256 + threadIdx.x) * 8;
  v8bf o;
#pragma unroll
  for (int j = 0; j < 8; ++j) o[j] = f2bf(in[i + j]);
  *(v8bf*)(out + i) = o;
}

// ---------------- pack row-major fp32 KxN weight into fragment-tiled bf16 ----------------
// Block (nt, kt) = 32x16 KxN tile; slot = (nt*(K/32)+kt)*32 + lane; 16 contiguous bf16 per slot
// in exactly B-fragment element order (lane -> col nt*16+(lane&15), K base kt*32+(lane>>4)*8).
__global__ __launch_bounds__(256) void pack_wfrag_kernel(const float* __restrict__ W,
                                                         __bf16* __restrict__ Wp,
                                                         int K, int N) {
  const int slot = blockIdx.x * 256 + threadIdx.x;
  const int lane = slot & 31;
  const int blk  = slot >> 5;
  const int kblocks = K >> 5;
  const int kt = blk % kblocks;
  const int nt = blk / kblocks;
  const int col = nt * 16 + (lane & 15);
  const int kb  = kt * 32 + ((lane >> 4) * 8);
  v16bf r;
#pragma unroll
  for (int i = 0; i < 8; ++i) r[i] = f2bf(W[(size_t)(kb + i) * N + col]);
#pragma unroll
  for (int i = 0; i < 8; ++i) r[8 + i] = f2bf(W[(size_t)(kb + 16 + i) * N + col]);
  *(v16bf*)(Wp + (size_t)slot * 16) = r;
}

// ---------------- per-head 64x64 projection (bf16 in, packed weight), row-major bf16 out ----------------
__global__ __launch_bounds__(128) void proj_head_rm_kernel(const __bf16* __restrict__ in,
                                                           const __bf16* __restrict__ Wp,
                                                           __bf16* __restrict__ out) {
  const int lane = threadIdx.x & 31;
  const int wid  = (blockIdx.x * blockDim.x + threadIdx.x) >> 5;
  const int head = wid & (cH - 1);
  const int rt   = wid >> 4;
  const __bf16* abase = in  + (size_t)rt * 16 * cD + head * cHD;
  __bf16*       obase = out + (size_t)rt * 16 * cD + head * cHD;
  v8f acc[4] = {};
#pragma unroll
  for (int kt = 0; kt < 2; ++kt) {
    v16bf a = load_afrag_bf16(abase + kt * 32, cD, lane);
#pragma unroll
    for (int c = 0; c < 4; ++c) {
      v16bf b = *(const v16bf*)(Wp + ((size_t)(c * 2 + kt) * 32 + lane) * 16);
      acc[c] = wmma_bf16(a, b, acc[c]);
    }
  }
  const int rowbase = (lane >> 4) * 8;
  const int lcol    = lane & 15;
#pragma unroll
  for (int c = 0; c < 4; ++c)
#pragma unroll
    for (int g = 0; g < 8; ++g)
      obase[(size_t)(rowbase + g) * cD + c * 16 + lcol] = f2bf(acc[c][g]);
}

// ---------------- per-head projection with transposed output: outT[(n*H+h)*HD+d][s] ----------------
__global__ __launch_bounds__(128) void proj_head_tr_kernel(const __bf16* __restrict__ in,
                                                           const __bf16* __restrict__ Wp,
                                                           __bf16* __restrict__ outT) {
  const int lane = threadIdx.x & 31;
  const int wid  = (blockIdx.x * blockDim.x + threadIdx.x) >> 5;
  const int head = wid & (cH - 1);
  const int rt   = wid >> 4;
  const int n    = rt >> 6;
  const int s0   = (rt & 63) * 16;
  const __bf16* abase = in + (size_t)rt * 16 * cD + head * cHD;
  v8f acc[4] = {};
#pragma unroll
  for (int kt = 0; kt < 2; ++kt) {
    v16bf a = load_afrag_bf16(abase + kt * 32, cD, lane);
#pragma unroll
    for (int c = 0; c < 4; ++c) {
      v16bf b = *(const v16bf*)(Wp + ((size_t)(c * 2 + kt) * 32 + lane) * 16);
      acc[c] = wmma_bf16(a, b, acc[c]);
    }
  }
  const int rowbase = (lane >> 4) * 8;
  const int lcol    = lane & 15;
#pragma unroll
  for (int c = 0; c < 4; ++c) {
    v8bf vs;
#pragma unroll
    for (int g = 0; g < 8; ++g) vs[g] = f2bf(acc[c][g]);
    const size_t chan = (size_t)(n * cH + head) * cHD + c * 16 + lcol;
    *(v8bf*)(outT + chan * cS + s0 + rowbase) = vs;  // 8 consecutive tokens, 16B store
  }
}

// ---------------- GEMM + bias, bf16 A x packed bf16 B -> fp32 C; wave tile 32x64, SW-pipelined ----------------
__global__ __launch_bounds__(128) void gemm_bias_kernel(const __bf16* __restrict__ A,
                                                        const __bf16* __restrict__ Bp,
                                                        const float* __restrict__ bias,
                                                        float* __restrict__ C,
                                                        int M, int N, int K) {
  const int lane   = threadIdx.x & 31;
  const int wid    = (blockIdx.x * blockDim.x + threadIdx.x) >> 5;
  const int ntiles = N >> 6;
  const int nt     = wid % ntiles;   // waves in a block share mt -> A stays cache-hot
  const int mt     = wid / ntiles;
  const int kblocks = K >> 5;
  const __bf16* a0 = A + (size_t)mt * 32 * K;
  const __bf16* a1 = a0 + (size_t)16 * K;
  const __bf16* bbase = Bp + (size_t)nt * 4 * kblocks * 32 * 16;
  v8f acc0[4] = {}, acc1[4] = {};

  // Register double-buffer: current k-step fragments
  v16bf a0c = load_afrag_bf16(a0, K, lane);
  v16bf a1c = load_afrag_bf16(a1, K, lane);
  v16bf bc[4];
#pragma unroll
  for (int c = 0; c < 4; ++c)
    bc[c] = *(const v16bf*)(bbase + (((size_t)c * kblocks) * 32 + lane) * 16);

  for (int kt = 0; kt < kblocks - 1; ++kt) {
    // issue next iteration's loads first; WMMAs below wait only on iteration-old loads
    const int ktn = kt + 1;
    v16bf a0n = load_afrag_bf16(a0 + ktn * 32, K, lane);
    v16bf a1n = load_afrag_bf16(a1 + ktn * 32, K, lane);
    v16bf bn[4];
#pragma unroll
    for (int c = 0; c < 4; ++c)
      bn[c] = *(const v16bf*)(bbase + (((size_t)c * kblocks + ktn) * 32 + lane) * 16);
    if (kt + 8 < kblocks)
      __builtin_prefetch(bbase + ((size_t)kt + 8) * 32 * 16 + lane * 16, 0, 1);
#pragma unroll
    for (int c = 0; c < 4; ++c) {
      acc0[c] = wmma_bf16(a0c, bc[c], acc0[c]);
      acc1[c] = wmma_bf16(a1c, bc[c], acc1[c]);
    }
    a0c = a0n;
    a1c = a1n;
#pragma unroll
    for (int c = 0; c < 4; ++c) bc[c] = bn[c];
  }
#pragma unroll
  for (int c = 0; c < 4; ++c) {
    acc0[c] = wmma_bf16(a0c, bc[c], acc0[c]);
    acc1[c] = wmma_bf16(a1c, bc[c], acc1[c]);
  }

  const int rowbase = (lane >> 4) * 8;
  const int lcol    = lane & 15;
#pragma unroll
  for (int c = 0; c < 4; ++c) {
    const int col  = nt * 64 + c * 16 + lcol;
    const float bb = bias[col];
#pragma unroll
    for (int g = 0; g < 8; ++g) {
      C[(size_t)(mt * 32 + rowbase + g) * N + col]      = acc0[c][g] + bb;
      C[(size_t)(mt * 32 + 16 + rowbase + g) * N + col] = acc1[c][g] + bb;
    }
  }
}

// ---------------- fused SwiGLU GEMM: G = silu(A@W12a+b1)*(A@W12b+b2); 32x32 tile, SW-pipelined ----------------
__global__ __launch_bounds__(128) void swiglu_gemm_kernel(const __bf16* __restrict__ A,
                                                          const __bf16* __restrict__ W12p,
                                                          const float* __restrict__ b12,
                                                          __bf16* __restrict__ G) {
  const int lane    = threadIdx.x & 31;
  const int wid     = (blockIdx.x * blockDim.x + threadIdx.x) >> 5;
  const int ntiles  = cHID / 32;  // 128
  const int nt      = wid % ntiles;
  const int mt      = wid / ntiles;
  const int kblocks = cD >> 5;    // 32
  const __bf16* a0 = A + (size_t)mt * 32 * cD;
  const __bf16* a1 = a0 + (size_t)16 * cD;
  // packed w12 has 512 column subtiles: x1 subtiles nt*2+c, x2 subtiles 256+nt*2+c
  const __bf16* b1base = W12p + (size_t)(nt * 2) * kblocks * 32 * 16;
  const __bf16* b2base = W12p + (size_t)(256 + nt * 2) * kblocks * 32 * 16;
  v8f p0[2] = {}, p1[2] = {}, q0[2] = {}, q1[2] = {};

  v16bf a0c = load_afrag_bf16(a0, cD, lane);
  v16bf a1c = load_afrag_bf16(a1, cD, lane);
  v16bf b1c[2], b2c[2];
#pragma unroll
  for (int c = 0; c < 2; ++c) {
    b1c[c] = *(const v16bf*)(b1base + (((size_t)c * kblocks) * 32 + lane) * 16);
    b2c[c] = *(const v16bf*)(b2base + (((size_t)c * kblocks) * 32 + lane) * 16);
  }

  for (int kt = 0; kt < kblocks - 1; ++kt) {
    const int ktn = kt + 1;
    v16bf a0n = load_afrag_bf16(a0 + ktn * 32, cD, lane);
    v16bf a1n = load_afrag_bf16(a1 + ktn * 32, cD, lane);
    v16bf b1n[2], b2n[2];
#pragma unroll
    for (int c = 0; c < 2; ++c) {
      b1n[c] = *(const v16bf*)(b1base + (((size_t)c * kblocks + ktn) * 32 + lane) * 16);
      b2n[c] = *(const v16bf*)(b2base + (((size_t)c * kblocks + ktn) * 32 + lane) * 16);
    }
    if (kt + 8 < kblocks) {
      __builtin_prefetch(b1base + ((size_t)kt + 8) * 32 * 16 + lane * 16, 0, 1);
      __builtin_prefetch(b2base + ((size_t)kt + 8) * 32 * 16 + lane * 16, 0, 1);
    }
#pragma unroll
    for (int c = 0; c < 2; ++c) {
      p0[c] = wmma_bf16(a0c, b1c[c], p0[c]);
      p1[c] = wmma_bf16(a1c, b1c[c], p1[c]);
      q0[c] = wmma_bf16(a0c, b2c[c], q0[c]);
      q1[c] = wmma_bf16(a1c, b2c[c], q1[c]);
    }
    a0c = a0n;
    a1c = a1n;
#pragma unroll
    for (int c = 0; c < 2; ++c) { b1c[c] = b1n[c]; b2c[c] = b2n[c]; }
  }
#pragma unroll
  for (int c = 0; c < 2; ++c) {
    p0[c] = wmma_bf16(a0c, b1c[c], p0[c]);
    p1[c] = wmma_bf16(a1c, b1c[c], p1[c]);
    q0[c] = wmma_bf16(a0c, b2c[c], q0[c]);
    q1[c] = wmma_bf16(a1c, b2c[c], q1[c]);
  }

  const int rowbase = (lane >> 4) * 8;
  const int lcol    = lane & 15;
#pragma unroll
  for (int c = 0; c < 2; ++c) {
    const int col   = nt * 32 + c * 16 + lcol;
    const float bb1 = b12[col];
    const float bb2 = b12[cHID + col];
#pragma unroll
    for (int g = 0; g < 8; ++g) {
      float x1 = p0[c][g] + bb1;
      float x2 = q0[c][g] + bb2;
      G[(size_t)(mt * 32 + rowbase + g) * cHID + col] =
          f2bf(x1 * (1.0f / (1.0f + __expf(-x1))) * x2);
      x1 = p1[c][g] + bb1;
      x2 = q1[c][g] + bb2;
      G[(size_t)(mt * 32 + 16 + rowbase + g) * cHID + col] =
          f2bf(x1 * (1.0f / (1.0f + __expf(-x1))) * x2);
    }
  }
}

// ---------------- flash attention (bf16 Q/K row-major, V transposed), ALiBi + key mask ----------------
__global__ __launch_bounds__(128) void attn_fa_kernel(const __bf16* __restrict__ Qm,
                                                      const __bf16* __restrict__ Km,
                                                      const __bf16* __restrict__ Vt,
                                                      const int* __restrict__ mask,
                                                      __bf16* __restrict__ Om) {
  __shared__ float pshare[4][16 * 32];
  const int lane    = threadIdx.x & 31;
  const int wslot   = threadIdx.x >> 5;
  const int wid     = blockIdx.x * 4 + wslot;
  const int qt      = wid & (cS / 16 - 1);
  const int h       = (wid >> 6) & (cH - 1);
  const int n       = wid >> 10;
  const int q0      = qt * 16;
  const int half    = lane >> 4;
  const int lcol    = lane & 15;
  const int rowbase = half * 8;

  const __bf16* qbase = Qm + ((size_t)n * cS + q0) * cD + h * cHD;
  v16bf qa0 = load_afrag_bf16(qbase, cD, lane);
  v16bf qa1 = load_afrag_bf16(qbase + 32, cD, lane);

  const float slope = exp2f(-(float)(h + 1));
  const float scale = 0.03125f;  // 1/sqrt(D=1024)

  float m_r[8], l_r[8];
#pragma unroll
  for (int g = 0; g < 8; ++g) { m_r[g] = -3.0e38f; l_r[g] = 0.0f; }
  v8f oacc[4] = {};
  float* myp = &pshare[wslot][0];

  for (int kt = 0; kt < cS; kt += 32) {
    v8f st[2] = {};
#pragma unroll
    for (int t = 0; t < 2; ++t) {
      const __bf16* kbase = Km + ((size_t)n * cS + kt + t * 16) * cD + h * cHD;
      st[t] = wmma_bf16(qa0, load_afrag_bf16(kbase, cD, lane), st[t]);
      st[t] = wmma_bf16(qa1, load_afrag_bf16(kbase + 32, cD, lane), st[t]);
    }
    // hoist V fragment loads: independent of P, overlap with softmax + LDS roundtrip
    v16bf vf[4];
#pragma unroll
    for (int c = 0; c < 4; ++c) {
      const __bf16* vp = Vt + ((size_t)(n * cH + h) * cHD + c * 16 + lcol) * cS + kt + half * 8;
      vf[c] = cat8(*(const v8bf*)vp, *(const v8bf*)(vp + 16));
    }
    float pm[2][8];
#pragma unroll
    for (int t = 0; t < 2; ++t) {
      const int kidx = kt + t * 16 + lcol;
      const int mk   = mask[n * cS + kidx];
#pragma unroll
      for (int g = 0; g < 8; ++g) {
        float e = st[t][g] - fabsf((float)(q0 + rowbase + g) - (float)kidx) * slope;
        e = (mk == 0) ? -1.0e4f : e;
        pm[t][g] = e * scale;
      }
    }
    float corr[8];
#pragma unroll
    for (int g = 0; g < 8; ++g) {
      float mx = fmaxf(pm[0][g], pm[1][g]);
#pragma unroll
      for (int off = 1; off < 16; off <<= 1) mx = fmaxf(mx, __shfl_xor(mx, off, 32));
      float mnew = fmaxf(m_r[g], mx);
      corr[g] = __expf(m_r[g] - mnew);
      float pe0 = __expf(pm[0][g] - mnew);
      float pe1 = __expf(pm[1][g] - mnew);
      pm[0][g] = pe0;
      pm[1][g] = pe1;
      float rs = pe0 + pe1;
#pragma unroll
      for (int off = 1; off < 16; off <<= 1) rs += __shfl_xor(rs, off, 32);
      l_r[g] = l_r[g] * corr[g] + rs;
      m_r[g] = mnew;
    }
#pragma unroll
    for (int c = 0; c < 4; ++c)
#pragma unroll
      for (int g = 0; g < 8; ++g) oacc[c][g] *= corr[g];

    // P (C-layout) -> LDS -> A-fragment
#pragma unroll
    for (int t = 0; t < 2; ++t)
#pragma unroll
      for (int g = 0; g < 8; ++g)
        myp[(rowbase + g) * 32 + t * 16 + lcol] = pm[t][g];
    __syncthreads();
    v16bf pa;
    {
      const float* pr = &myp[lcol * 32 + half * 8];
#pragma unroll
      for (int i = 0; i < 8; ++i) pa[i] = f2bf(pr[i]);
#pragma unroll
      for (int i = 0; i < 8; ++i) pa[8 + i] = f2bf(pr[16 + i]);
    }
    __syncthreads();

    // O += P(16x32) @ V(32x64)
#pragma unroll
    for (int c = 0; c < 4; ++c) oacc[c] = wmma_bf16(pa, vf[c], oacc[c]);
  }
#pragma unroll
  for (int c = 0; c < 4; ++c)
#pragma unroll
    for (int g = 0; g < 8; ++g)
      Om[((size_t)n * cS + q0 + rowbase + g) * cD + h * cHD + c * 16 + lcol] =
          f2bf(oacc[c][g] / l_r[g]);
}

// ---------------- LayerNorm (fp32 in): bf16-out and f32-out variants ----------------
template <typename OutT>
__device__ __forceinline__ void layernorm_body(const float* __restrict__ x,
                                               const float* __restrict__ w,
                                               const float* __restrict__ b,
                                               OutT* __restrict__ o) {
  __shared__ float s1[256], s2[256];
  const int row = blockIdx.x;
  const int t   = threadIdx.x;
  const float* xr = x + (size_t)row * cD;
  float v[4];
  float sum = 0.0f, sq = 0.0f;
#pragma unroll
  for (int i = 0; i < 4; ++i) {
    v[i] = xr[t + i * 256];
    sum += v[i];
    sq  += v[i] * v[i];
  }
  s1[t] = sum;
  s2[t] = sq;
  __syncthreads();
  for (int off = 128; off > 0; off >>= 1) {
    if (t < off) { s1[t] += s1[t + off]; s2[t] += s2[t + off]; }
    __syncthreads();
  }
  const float mu  = s1[0] * (1.0f / cD);
  const float var = s2[0] * (1.0f / cD) - mu * mu;
  const float rs  = rsqrtf(var + 1e-5f);
  OutT* orow = o + (size_t)row * cD;
#pragma unroll
  for (int i = 0; i < 4; ++i) {
    const int c = t + i * 256;
    orow[c] = (OutT)((v[i] - mu) * rs * w[c] + b[c]);
  }
}

__global__ __launch_bounds__(256) void layernorm_bf16_kernel(const float* __restrict__ x,
                                                             const float* __restrict__ w,
                                                             const float* __restrict__ b,
                                                             __bf16* __restrict__ o) {
  layernorm_body<__bf16>(x, w, b, o);
}

__global__ __launch_bounds__(256) void layernorm_f32_kernel(const float* __restrict__ x,
                                                            const float* __restrict__ w,
                                                            const float* __restrict__ b,
                                                            float* __restrict__ o) {
  layernorm_body<float>(x, w, b, o);
}

extern "C" void kernel_launch(void* const* d_in, const int* in_sizes, int n_in,
                              void* d_out, int out_size, void* d_ws, size_t ws_size,
                              hipStream_t stream) {
  (void)in_sizes; (void)n_in; (void)out_size; (void)ws_size;
  const float* x        = (const float*)d_in[0];
  const float* value    = (const float*)d_in[1];
  const float* keyi     = (const float*)d_in[2];
  const int*   src_mask = (const int*)d_in[3];
  const int*   tgt_mask = (const int*)d_in[4];
  const float* sa_wv = (const float*)d_in[5];
  const float* sa_wk = (const float*)d_in[6];
  const float* sa_wq = (const float*)d_in[7];
  const float* sa_wo = (const float*)d_in[8];
  const float* sa_bo = (const float*)d_in[9];
  const float* ca_wv = (const float*)d_in[10];
  const float* ca_wk = (const float*)d_in[11];
  const float* ca_wq = (const float*)d_in[12];
  const float* ca_wo = (const float*)d_in[13];
  const float* ca_bo = (const float*)d_in[14];
  const float* ln1_w = (const float*)d_in[15];
  const float* ln1_b = (const float*)d_in[16];
  const float* ln2_w = (const float*)d_in[17];
  const float* ln2_b = (const float*)d_in[18];
  const float* ln3_w = (const float*)d_in[19];
  const float* ln3_b = (const float*)d_in[20];
  const float* w12   = (const float*)d_in[21];
  const float* b12   = (const float*)d_in[22];
  const float* w3    = (const float*)d_in[23];
  const float* b3    = (const float*)d_in[24];

  const size_t NSD = (size_t)cN * cS * cD;  // 4,194,304 elements
  __bf16* p = (__bf16*)d_ws;
  __bf16* xb   = p; p += NSD;
  __bf16* valb = p; p += NSD;
  __bf16* keyb = p; p += NSD;
  __bf16* Qb   = p; p += NSD;
  __bf16* Kb   = p; p += NSD;
  __bf16* Vt   = p; p += NSD;   // (n,h,d,S)
  __bf16* AOb  = p; p += NSD;
  __bf16* H1b  = p; p += NSD;
  __bf16* H2b  = p; p += NSD;
  __bf16* Gb   = p; p += (size_t)cN * cS * cHID;
  __bf16* Pwq  = p; p += 64 * 64;
  __bf16* Pwk  = p; p += 64 * 64;
  __bf16* Pwv  = p; p += 64 * 64;
  __bf16* Pcwq = p; p += 64 * 64;
  __bf16* Pcwk = p; p += 64 * 64;
  __bf16* Pcwv = p; p += 64 * 64;
  __bf16* Pwo  = p; p += (size_t)cD * cD;
  __bf16* Pcwo = p; p += (size_t)cD * cD;
  __bf16* Pw12 = p; p += (size_t)cD * 2 * cHID;
  __bf16* Pw3  = p; p += (size_t)cHID * cD;
  float*  Ff   = (float*)p;  // fp32 GEMM output scratch, reused

  const dim3 blk(128);
  const int rows        = cN * cS;                       // 4096
  const int cvt_blocks  = (int)(NSD / 2048);             // 2048
  const int proj_blocks = (rows / 16) * cH / 4;          // 1024
  const int attn_blocks = cN * cH * (cS / 16) / 4;       // 1024
  const int gemm_blocks = (rows / 32) * (cD / 64) / 4;   // 512
  const int swi_blocks  = (rows / 32) * (cHID / 32) / 4; // 4096

  // ---- one-time packing / conversion passes ----
  cvt_bf16_kernel<<<cvt_blocks, 256, 0, stream>>>(x, xb);
  cvt_bf16_kernel<<<cvt_blocks, 256, 0, stream>>>(value, valb);
  cvt_bf16_kernel<<<cvt_blocks, 256, 0, stream>>>(keyi, keyb);
  pack_wfrag_kernel<<<1, 256, 0, stream>>>(sa_wq, Pwq, cHD, cHD);
  pack_wfrag_kernel<<<1, 256, 0, stream>>>(sa_wk, Pwk, cHD, cHD);
  pack_wfrag_kernel<<<1, 256, 0, stream>>>(sa_wv, Pwv, cHD, cHD);
  pack_wfrag_kernel<<<1, 256, 0, stream>>>(ca_wq, Pcwq, cHD, cHD);
  pack_wfrag_kernel<<<1, 256, 0, stream>>>(ca_wk, Pcwk, cHD, cHD);
  pack_wfrag_kernel<<<1, 256, 0, stream>>>(ca_wv, Pcwv, cHD, cHD);
  pack_wfrag_kernel<<<256, 256, 0, stream>>>(sa_wo, Pwo, cD, cD);
  pack_wfrag_kernel<<<256, 256, 0, stream>>>(ca_wo, Pcwo, cD, cD);
  pack_wfrag_kernel<<<2048, 256, 0, stream>>>(w12, Pw12, cD, 2 * cHID);
  pack_wfrag_kernel<<<1024, 256, 0, stream>>>(w3, Pw3, cHID, cD);

  // ---- self attention: mha(values=x, keys=x, queries=x, tgt_mask) ----
  proj_head_rm_kernel<<<proj_blocks, blk, 0, stream>>>(xb, Pwq, Qb);
  proj_head_rm_kernel<<<proj_blocks, blk, 0, stream>>>(xb, Pwk, Kb);
  proj_head_tr_kernel<<<proj_blocks, blk, 0, stream>>>(xb, Pwv, Vt);
  attn_fa_kernel<<<attn_blocks, blk, 0, stream>>>(Qb, Kb, Vt, tgt_mask, AOb);
  gemm_bias_kernel<<<gemm_blocks, blk, 0, stream>>>(AOb, Pwo, sa_bo, Ff, rows, cD, cD);
  layernorm_bf16_kernel<<<rows, 256, 0, stream>>>(Ff, ln1_w, ln1_b, H1b);

  // ---- cross attention: mha(values=h, keys=key_in, queries=value_in, src_mask) ----
  proj_head_rm_kernel<<<proj_blocks, blk, 0, stream>>>(valb, Pcwq, Qb);
  proj_head_rm_kernel<<<proj_blocks, blk, 0, stream>>>(keyb, Pcwk, Kb);
  proj_head_tr_kernel<<<proj_blocks, blk, 0, stream>>>(H1b, Pcwv, Vt);
  attn_fa_kernel<<<attn_blocks, blk, 0, stream>>>(Qb, Kb, Vt, src_mask, AOb);
  gemm_bias_kernel<<<gemm_blocks, blk, 0, stream>>>(AOb, Pcwo, ca_bo, Ff, rows, cD, cD);
  layernorm_bf16_kernel<<<rows, 256, 0, stream>>>(Ff, ln2_w, ln2_b, H2b);

  // ---- SwiGLU FFN + final LN ----
  swiglu_gemm_kernel<<<swi_blocks, blk, 0, stream>>>(H2b, Pw12, b12, Gb);
  gemm_bias_kernel<<<gemm_blocks, blk, 0, stream>>>(Gb, Pw3, b3, Ff, rows, cD, cHID);
  layernorm_f32_kernel<<<rows, 256, 0, stream>>>(Ff, ln3_w, ln3_b, (float*)d_out);
}
